// InfoNCELoss_78434692759648
// MI455X (gfx1250) — compile-verified, matrix operand order checked by reading, not run on hardware
//
#include <hip/hip_runtime.h>
#include <hip/hip_bf16.h>
#include <math.h>

// ---------------- problem constants ----------------
#define DDIM   256            // feature dim (fixed by reference)
#define LDSTR  264            // padded LDS row stride in bf16 elems (+16B pad -> conflict-free b128 reads)
#define SPLITS 32             // key-dimension splits across blockIdx.y
#define TEMP       0.07f
#define LOG2E      1.4426950408889634f
#define EPS_LOSS   1e-8f
#define NORM_EPS   1e-12f

typedef __attribute__((ext_vector_type(8)))  __bf16 v8bf;
typedef __attribute__((ext_vector_type(16))) __bf16 v16bf;
typedef __attribute__((ext_vector_type(8)))  float  v8f;

typedef __attribute__((address_space(3))) __bf16 lds_bf16;

static __device__ __forceinline__ v16bf cat16(v8bf lo, v8bf hi) {
  return __builtin_shufflevector(lo, hi, 0,1,2,3,4,5,6,7,8,9,10,11,12,13,14,15);
}

// CDNA5 async global->LDS copy (16 bytes per lane), tracked by ASYNCcnt.
static __device__ __forceinline__ void async_b128_to_lds(__bf16* dstGeneric,
                                                         const __bf16* src) {
  lds_bf16* d3 = (lds_bf16*)dstGeneric;           // addrspacecast -> 32-bit LDS addr
  asm volatile("global_load_async_to_lds_b128 %0, %1, off"
               :: "v"(d3), "v"(src)
               : "memory");
}
static __device__ __forceinline__ void wait_async0() {
  asm volatile("s_wait_asynccnt 0x0" ::: "memory");
}

// ---------------------------------------------------------------------------
// Kernel 1: L2-normalize rows of Q and K, convert to bf16.
// Queries additionally get the fused scale log2(e)/T so the GEMM accumulator
// is directly in log2-domain (epilogue = one v_exp_f32 per element).
// One 256-thread block per row (D == 256).
// ---------------------------------------------------------------------------
__global__ __launch_bounds__(256)
void norm_cvt_kernel(const float* __restrict__ q, const float* __restrict__ k,
                     __bf16* __restrict__ qb, __bf16* __restrict__ kb,
                     int N, float qscale) {
  const int row = blockIdx.x;
  const int tid = threadIdx.x;

  const float* src;
  __bf16* dst;
  float extra;
  if (row < N) {
    src = q + (size_t)row * DDIM;
    dst = qb + (size_t)row * DDIM;
    extra = qscale;                       // log2(e)/T folded into queries
  } else {
    const int r2 = row - N;
    src = k + (size_t)r2 * DDIM;
    dst = kb + (size_t)r2 * DDIM;
    extra = 1.0f;
  }

  float x = src[tid];
  float ss = x * x;
#pragma unroll
  for (int off = 16; off; off >>= 1) ss += __shfl_xor(ss, off, 32);

  __shared__ float warp_ss[8];
  __shared__ float total_ss;
  if ((tid & 31) == 0) warp_ss[tid >> 5] = ss;
  __syncthreads();
  if (tid == 0) {
    float t = 0.f;
#pragma unroll
    for (int i = 0; i < 8; ++i) t += warp_ss[i];
    total_ss = t;
  }
  __syncthreads();

  const float nrm = sqrtf(total_ss);
  const float inv = extra / fmaxf(nrm, NORM_EPS);   // torch F.normalize semantics
  dst[tid] = (__bf16)(x * inv);
}

// ---------------------------------------------------------------------------
// Kernel 2: zero the pos/neg partial-sum arrays (atomically accumulated).
// ---------------------------------------------------------------------------
__global__ __launch_bounds__(256)
void zero_kernel(float* __restrict__ p, int n) {
  const int i = blockIdx.x * 256 + threadIdx.x;
  if (i < n) p[i] = 0.f;
}

// ---------------------------------------------------------------------------
// Kernel 3: fused  exp2( (q/|q|)·(k/|k|) · log2e/T )  with masked row sums.
//   block = 256 threads = 8 waves; block owns 256 query rows:
//     wave w owns row tiles [bx*256 + w*16]  and  [bx*256 + 128 + w*16]
//   -> every B fragment fetched from LDS feeds TWO WMMAs (wmma:ds = 1:1).
//   blockIdx.y selects a contiguous slice of keys (M / SPLITS keys).
//   B tiles (16 keys x 256 dims, bf16) double-buffered in LDS, filled with
//   CDNA5 async global->LDS copies (ASYNCcnt), consumed via ds_load_b128.
//   A fragments (2 x 16 rows x 256 dims) live in registers.
// Fragment addressing per CDNA5 ISA (16-bit A 16x32): lane group g = lane>>4
// holds dims [32k + 8g, +8) and [32k + 16 + 8g, +8); B is symmetric since
// B = K^T and K is row-major.  MC: compile-time M (0 = runtime) so the 16
// mask loads use immediate offsets (r*M*4 fits signed 24-bit IOFFSET).
// ---------------------------------------------------------------------------
template <int MC>
__global__ __launch_bounds__(256)
void infonce_gemm_kernel(const __bf16* __restrict__ qb,
                         const __bf16* __restrict__ kb,
                         const int* __restrict__ pmask,
                         float* __restrict__ ps, float* __restrict__ ns,
                         int Mrt, int tilesPerWG) {
  const size_t M = MC ? (size_t)MC : (size_t)Mrt;

  __shared__ __bf16 ldsB[2][16 * LDSTR];

  const int tid    = threadIdx.x;
  const int lane   = tid & 31;
  const int wave   = tid >> 5;          // 0..7
  const int g      = lane >> 4;         // half-wave group (0/1)
  const int n16    = lane & 15;         // fragment row/col within tile
  const int rowT0  = blockIdx.x * 256 + wave * 16;        // row tile 0
  const int rowT1  = rowT0 + 128;                          // row tile 1
  const long keyBase0 = (long)blockIdx.y * tilesPerWG * 16;

  // ---- preload A fragments for both row tiles, all 8 K-steps ----
  v8bf a0Lo[8], a0Hi[8], a1Lo[8], a1Hi[8];
  {
    const __bf16* qrow0 = qb + (size_t)(rowT0 + n16) * DDIM + g * 8;
    const __bf16* qrow1 = qb + (size_t)(rowT1 + n16) * DDIM + g * 8;
#pragma unroll
    for (int ks = 0; ks < 8; ++ks) {
      a0Lo[ks] = *(const v8bf*)(qrow0 + ks * 32);
      a0Hi[ks] = *(const v8bf*)(qrow0 + ks * 32 + 16);
      a1Lo[ks] = *(const v8bf*)(qrow1 + ks * 32);
      a1Hi[ks] = *(const v8bf*)(qrow1 + ks * 32 + 16);
    }
  }

  // per-lane mask bases: row = rowT + r + 8g, col = keyBase0 + j*16 + n16
  const int* pm0 = pmask + (size_t)(rowT0 + 8 * g) * M + (size_t)(keyBase0 + n16);
  const int* pm1 = pmask + (size_t)(rowT1 + 8 * g) * M + (size_t)(keyBase0 + n16);

  // ---- async-stage B tile 0 into LDS buffer 0 (32B per thread) ----
  const int kk = tid >> 4;              // key row within tile   (0..15)
  const int cc = tid & 15;              // 16-elem chunk along D (0..15)
  {
    const __bf16* src = kb + (size_t)(keyBase0 + kk) * DDIM + cc * 16;
    __bf16* dst = &ldsB[0][kk * LDSTR + cc * 16];
    async_b128_to_lds(dst, src);
    async_b128_to_lds(dst + 8, src + 8);
  }
  wait_async0();
  __syncthreads();

  float pos0[8], neg0[8], pos1[8], neg1[8];
#pragma unroll
  for (int r = 0; r < 8; ++r) {
    pos0[r] = 0.f; neg0[r] = 0.f; pos1[r] = 0.f; neg1[r] = 0.f;
  }

  for (int j = 0; j < tilesPerWG; ++j) {
    const int cur  = j & 1;
    const bool more = (j + 1) < tilesPerWG;

    // kick off async copy of the next B tile into the other buffer
    if (more) {
      const __bf16* src =
          kb + (size_t)(keyBase0 + (size_t)(j + 1) * 16 + kk) * DDIM + cc * 16;
      __bf16* dst = &ldsB[cur ^ 1][kk * LDSTR + cc * 16];
      async_b128_to_lds(dst, src);
      async_b128_to_lds(dst + 8, src + 8);
    }

    // mask values for both 16x16 tiles (immediate-offset loads off two bases)
    const int* pm0j = pm0 + (size_t)j * 16;
    const int* pm1j = pm1 + (size_t)j * 16;
    int mv0[8], mv1[8];
#pragma unroll
    for (int r = 0; r < 8; ++r) {
      mv0[r] = pm0j[(size_t)r * M];
      mv1[r] = pm1j[(size_t)r * M];
    }

    // each B fragment from LDS feeds two WMMAs (row tile 0 and 1)
    v8f acc0 = {}, acc1 = {};
    const __bf16* brow = &ldsB[cur][n16 * LDSTR + g * 8];
#pragma unroll
    for (int ks = 0; ks < 8; ++ks) {
      v8bf bLo = *(const v8bf*)(brow + ks * 32);
      v8bf bHi = *(const v8bf*)(brow + ks * 32 + 16);
      const v16bf b = cat16(bLo, bHi);
      acc0 = __builtin_amdgcn_wmma_f32_16x16x32_bf16(
          false, cat16(a0Lo[ks], a0Hi[ks]), false, b, (short)0, acc0, false, false);
      acc1 = __builtin_amdgcn_wmma_f32_16x16x32_bf16(
          false, cat16(a1Lo[ks], a1Hi[ks]), false, b, (short)0, acc1, false, false);
    }

    // ---- epilogue: exp2 (scale pre-folded) + masked accumulation ----
#pragma unroll
    for (int r = 0; r < 8; ++r) {
      const float e0 = __builtin_amdgcn_exp2f(acc0[r]);
      const float e1 = __builtin_amdgcn_exp2f(acc1[r]);
      pos0[r] += mv0[r] ? e0 : 0.f;
      neg0[r] += mv0[r] ? 0.f : e0;
      pos1[r] += mv1[r] ? e1 : 0.f;
      neg1[r] += mv1[r] ? 0.f : e1;
    }

    // own async copies must complete before the barrier that precedes
    // the iteration which consumes them (cross-wave visibility via barrier)
    wait_async0();
    __syncthreads();
  }

  // ---- per-row reduction across the 16 lanes of each half-wave ----
#pragma unroll
  for (int r = 0; r < 8; ++r) {
    float p0 = pos0[r], q0 = neg0[r], p1 = pos1[r], q1 = neg1[r];
#pragma unroll
    for (int off = 8; off; off >>= 1) {
      p0 += __shfl_xor(p0, off, 32);
      q0 += __shfl_xor(q0, off, 32);
      p1 += __shfl_xor(p1, off, 32);
      q1 += __shfl_xor(q1, off, 32);
    }
    if (n16 == 0) {                      // lane 0 -> rows r, lane 16 -> rows r+8
      const int row0 = rowT0 + r + 8 * g;
      const int row1 = rowT1 + r + 8 * g;
      atomicAdd(&ps[row0], p0);
      atomicAdd(&ns[row0], q0);
      atomicAdd(&ps[row1], p1);
      atomicAdd(&ns[row1], q1);
    }
  }
}

// ---------------------------------------------------------------------------
// Kernel 4: loss = max( -mean(log((pos+eps)/(pos+neg+2eps))), 0 )
// ---------------------------------------------------------------------------
__global__ __launch_bounds__(256)
void finalize_kernel(const float* __restrict__ ps, const float* __restrict__ ns,
                     float* __restrict__ out, int N) {
  const int tid = threadIdx.x;
  float acc = 0.f;
  for (int i = tid; i < N; i += 256) {
    const float p = ps[i] + EPS_LOSS;
    const float q = ns[i] + EPS_LOSS;
    acc += -__logf(p / (p + q));
  }
#pragma unroll
  for (int off = 16; off; off >>= 1) acc += __shfl_xor(acc, off, 32);

  __shared__ float w[8];
  if ((tid & 31) == 0) w[tid >> 5] = acc;
  __syncthreads();
  if (tid == 0) {
    float t = 0.f;
#pragma unroll
    for (int i = 0; i < 8; ++i) t += w[i];
    out[0] = fmaxf(t / (float)N, 0.0f);
  }
}

// ---------------------------------------------------------------------------
extern "C" void kernel_launch(void* const* d_in, const int* in_sizes, int n_in,
                              void* d_out, int out_size, void* d_ws, size_t ws_size,
                              hipStream_t stream) {
  const float* q  = (const float*)d_in[0];
  const float* k  = (const float*)d_in[1];
  const int*   pm = (const int*)d_in[2];

  const int N = in_sizes[0] / DDIM;   // 4096
  const int M = in_sizes[1] / DDIM;   // 32768

  // workspace layout: qb (bf16) | kb (bf16) | pos[N] | neg[N]
  char* ws = (char*)d_ws;
  const size_t qbytes = ((size_t)N * DDIM * sizeof(__bf16) + 255) & ~(size_t)255;
  const size_t kbytes = ((size_t)M * DDIM * sizeof(__bf16) + 255) & ~(size_t)255;
  __bf16* qb = (__bf16*)ws;
  __bf16* kb = (__bf16*)(ws + qbytes);
  float*  ps = (float*)(ws + qbytes + kbytes);
  float*  ns = ps + N;
  (void)ws_size; (void)n_in; (void)out_size;

  // 1) normalize + convert (log2e/T folded into queries)
  norm_cvt_kernel<<<N + M, 256, 0, stream>>>(q, k, qb, kb, N, LOG2E / TEMP);

  // 2) zero accumulators
  zero_kernel<<<(2 * N + 255) / 256, 256, 0, stream>>>(ps, 2 * N);

  // 3) fused WMMA GEMM + exp + masked row sums
  const int tilesPerWG = M / (16 * SPLITS);
  dim3 grid(N / 256, SPLITS);
  if (M == 32768) {
    infonce_gemm_kernel<32768><<<grid, 256, 0, stream>>>(qb, kb, pm, ps, ns, M,
                                                         tilesPerWG);
  } else {
    infonce_gemm_kernel<0><<<grid, 256, 0, stream>>>(qb, kb, pm, ps, ns, M,
                                                     tilesPerWG);
  }

  // 4) final scalar loss
  finalize_kernel<<<1, 256, 0, stream>>>(ps, ns, (float*)d_out, N);
}
